// MultiHeadAttention_63823214019237
// MI455X (gfx1250) — compile-verified, matrix operand order checked by reading, not run on hardware
//
#include <hip/hip_runtime.h>
#include <hip/hip_bf16.h>

// ---- problem constants -------------------------------------------------
#define BATCH   2
#define SEQ     2048
#define DMODEL  1024
#define NHEAD   16
#define HDIM    64
#define MROWS   (BATCH * SEQ)          // 4096
#define QKV_ELEMS ((size_t)BATCH * NHEAD * SEQ * HDIM)   // 4,194,304
#define OUT0_ELEMS ((size_t)BATCH * SEQ * DMODEL)        // 4,194,304

typedef __attribute__((ext_vector_type(16))) _Float16 v16h;
typedef __attribute__((ext_vector_type(8)))  _Float16 v8h;
typedef __attribute__((ext_vector_type(8)))  float    v8f;
typedef __attribute__((ext_vector_type(4)))  float    v4f;
typedef __attribute__((ext_vector_type(4)))  int      v4i;

#define BSTR 40   // LDS row stride (halfs) for 32-wide k-slab: 80B, 16B aligned, bank-spread

#if defined(__has_builtin)
#if __has_builtin(__builtin_amdgcn_global_load_async_to_lds_b128)
#define HAVE_ASYNC_LDS 1
#endif
#endif

__device__ __forceinline__ void wait_asynccnt0() {
#if defined(__has_builtin) && __has_builtin(__builtin_amdgcn_s_wait_asynccnt)
    __builtin_amdgcn_s_wait_asynccnt(0);
#else
    asm volatile("s_wait_asynccnt 0" ::: "memory");
#endif
}

// D = A(16x32 f16) * B(32x16 f16) + C(f32)
__device__ __forceinline__ v8f wmma_f16(v16h a, v16h b, v8f c) {
    return __builtin_amdgcn_wmma_f32_16x16x32_f16(
        false, a, false, b, (short)0, c, false, false);
}

// WMMA 16-bit fragment from an f32 row. p = row + hl*8 (+kk by caller).
// K packing: e=0..7 -> k=hl*8+e ; e=8..15 -> k=16+hl*8+(e-8).
__device__ __forceinline__ v16h gather16_f32(const float* __restrict__ p) {
    const v4f* q = (const v4f*)p;
    v4f x0 = q[0], x1 = q[1], x2 = q[4], x3 = q[5];
    v16h r;
#pragma unroll
    for (int i = 0; i < 4; ++i) {
        r[i]      = (_Float16)x0[i];
        r[4 + i]  = (_Float16)x1[i];
        r[8 + i]  = (_Float16)x2[i];
        r[12 + i] = (_Float16)x3[i];
    }
    return r;
}

// Same fragment from an f16 row (two contiguous 16B chunks). Works for
// global or LDS pointers (addrspace inferred).
__device__ __forceinline__ v16h gather16_f16(const _Float16* p) {
    v8h a = *(const v8h*)p;
    v8h b = *(const v8h*)(p + 16);
    v16h r;
#pragma unroll
    for (int i = 0; i < 8; ++i) { r[i] = a[i]; r[8 + i] = b[i]; }
    return r;
}

// ======================================================================
// Kernel 1: projection  Y = (X @ W^T) * scale   (f32 in, f16 out)
//   MODE 0: Y layout [B,H,S,hd]   (q / k)     MODE 1: [B,H,hd,S] (vT)
// block 256 (8 waves): tile 128(M) x 64(N); W k-slab staged in LDS (x2 buf)
// ======================================================================
template <int MODE>
__global__ __launch_bounds__(256) void proj_kernel(
    const float* __restrict__ X, const float* __restrict__ W,
    _Float16* __restrict__ Y, float scale)
{
    __shared__ __align__(16) _Float16 sB[2][64 * BSTR];   // 2 x 5KB

    const int tid  = threadIdx.x;
    const int wave = tid >> 5, lane = tid & 31;
    const int lm = lane & 15, hl = lane >> 4;
    const int n0 = blockIdx.x * 64;
    const int m0 = blockIdx.y * 128 + wave * 16;

    // cooperative W-slab fetch: 64 rows x 32 f32 -> f16; 8 floats/thread
    const int wr = tid >> 2;                 // 0..63
    const int wc = (tid & 3) * 8;            // 0,8,16,24
    const float* wsrc = W + (size_t)(n0 + wr) * DMODEL + wc;
    v4f f0, f1;

    const float* arow = X + (size_t)(m0 + lm) * DMODEL + hl * 8;

    v8f acc[4] = {};

    // prologue: slab kk=0
    f0 = *(const v4f*)(wsrc);  f1 = *(const v4f*)(wsrc + 4);
    {
        v8h h;
#pragma unroll
        for (int i = 0; i < 4; ++i) { h[i] = (_Float16)f0[i]; h[4 + i] = (_Float16)f1[i]; }
        *(v8h*)&sB[0][wr * BSTR + wc] = h;
    }
    __syncthreads();

    v16h a_cur = gather16_f32(arow);

#pragma unroll 2
    for (int kk = 0; kk < DMODEL; kk += 32) {
        const int cur = (kk >> 5) & 1;
        const bool last = (kk + 32 >= DMODEL);
        if (!last) {                               // fetch next slab early
            f0 = *(const v4f*)(wsrc + kk + 32);
            f1 = *(const v4f*)(wsrc + kk + 36);
            __builtin_prefetch(arow + kk + 64, 0, 1);
        }
        v16h a_nxt = last ? a_cur : gather16_f32(arow + kk + 32);

        v16h bf[4];
#pragma unroll
        for (int t = 0; t < 4; ++t)
            bf[t] = gather16_f16(&sB[cur][(t * 16 + lm) * BSTR + hl * 8]);
#pragma unroll
        for (int t = 0; t < 4; ++t)
            acc[t] = wmma_f16(a_cur, bf[t], acc[t]);

        if (!last) {
            __syncthreads();                       // all waves done with buf cur^1
            v8h h;
#pragma unroll
            for (int i = 0; i < 4; ++i) { h[i] = (_Float16)f0[i]; h[4 + i] = (_Float16)f1[i]; }
            *(v8h*)&sB[cur ^ 1][wr * BSTR + wc] = h;
            __syncthreads();
        }
        a_cur = a_nxt;
    }

#pragma unroll
    for (int t = 0; t < 4; ++t)
#pragma unroll
        for (int r = 0; r < 8; ++r) {
            int m = m0 + r + 8 * hl;
            int n = n0 + t * 16 + lm;
            int bb = m >> 11, s = m & (SEQ - 1);
            int h = n >> 6, d = n & (HDIM - 1);
            size_t off = (MODE == 0)
                ? ((size_t)((bb * NHEAD + h) * SEQ + s)) * HDIM + d
                : ((size_t)((bb * NHEAD + h) * HDIM + d)) * SEQ + s;
            Y[off] = (_Float16)(acc[t][r] * scale);
        }
}

// ======================================================================
// Kernel 2: scores[b,h,q,k] = q16 . k16  (+mask), masked scores -> d_out
// grid (2, 128, 32); block 256.  q-tile staged in LDS (async path when
// available); K frags pipelined with constant-offset loads.
// ======================================================================
__global__ __launch_bounds__(256) void score_kernel(
    const _Float16* __restrict__ q16, const _Float16* __restrict__ k16,
    const int* __restrict__ mask, float* __restrict__ scores)
{
    __shared__ __align__(16) _Float16 sQ[16 * HDIM];     // 2KB

    const int tid  = threadIdx.x;
    const int wave = tid >> 5, lane = tid & 31;
    const int lm = lane & 15, hl = lane >> 4;
    const int bh = blockIdx.z;
    const int q0 = blockIdx.y * 16;
    const int kkb = blockIdx.x * 1024 + wave * 128;
    const int b = bh >> 4;

    const _Float16* qbase = q16 + (size_t)bh * SEQ * HDIM;

    // stage shared 16x64 q tile (all 8 waves use the same one)
    if (tid < 128) {
        int r = tid >> 3, c = (tid & 7) * 8;
        const _Float16* g = qbase + (size_t)(q0 + r) * HDIM + c;
#if defined(HAVE_ASYNC_LDS)
        __builtin_amdgcn_global_load_async_to_lds_b128(
            (__attribute__((address_space(1))) v4i*)g,
            (__attribute__((address_space(3))) v4i*)&sQ[r * HDIM + c],
            /*offset=*/0, /*cpol=*/0);
#else
        *(v8h*)&sQ[r * HDIM + c] = *(const v8h*)g;
#endif
    }
#if defined(HAVE_ASYNC_LDS)
    wait_asynccnt0();
#endif
    __syncthreads();

    // per-lane base for K rows; fragment (t, step) at constant offset
    const _Float16* kb = k16 + (size_t)bh * SEQ * HDIM
                       + (size_t)(kkb + lm) * HDIM + hl * 8;

    v16h a0 = gather16_f16(&sQ[lm * HDIM + 0  + hl * 8]);
    v16h a1 = gather16_f16(&sQ[lm * HDIM + 32 + hl * 8]);

    v8f acc[8] = {};
    v16h bcur = gather16_f16(kb);           // u=0: t=0, step=0
#pragma unroll
    for (int u = 0; u < 16; ++u) {
        const int t = u & 7, step = u >> 3;
        v16h bnxt = bcur;
        if (u < 15) {
            const int tn = (u + 1) & 7, sn = (u + 1) >> 3;
            bnxt = gather16_f16(kb + tn * 16 * HDIM + sn * 32);
        }
        acc[t] = wmma_f16(step ? a1 : a0, bcur, acc[t]);
        bcur = bnxt;
    }

#pragma unroll
    for (int t = 0; t < 8; ++t)
#pragma unroll
        for (int r = 0; r < 8; ++r) {
            int q = q0 + r + 8 * hl;
            int kc = kkb + t * 16 + lm;
            int mv = mask[((size_t)b * SEQ + q) * SEQ + kc];
            float v = acc[t][r];
            scores[((size_t)bh * SEQ + q) * SEQ + kc] = (mv == 0) ? -1e10f : v;
        }
}

// ======================================================================
// Kernel 3: softmax over score rows + P@V  ->  oh16 [B,S,D] f16
// grid (128, 32); block 256 (8 waves). One 16-row q-block per block.
// ======================================================================
__global__ __launch_bounds__(256) void softmax_pv_kernel(
    const float* __restrict__ scores, const _Float16* __restrict__ vT,
    _Float16* __restrict__ oh)
{
    __shared__ float red[16][17];
    __shared__ float rmax_s[16];
    __shared__ float rinv_s[16];
    __shared__ __align__(16) float partial[8][16][64];   // 32 KB

    const int tid = threadIdx.x;
    const int bh = blockIdx.y;
    const int q0 = blockIdx.x * 16;
    const int b = bh >> 4, h = bh & 15;

    // ---- phase A: per-row max and sum(exp) -----------------------------
    {
        const int row = tid >> 4, ci = tid & 15;
        const float* srow = scores + ((size_t)bh * SEQ + q0 + row) * SEQ;
        const v4f* s4 = (const v4f*)srow + ci * 32;      // 128 floats/thread
        float lmax = -3.0e38f;
#pragma unroll 8
        for (int i = 0; i < 32; ++i) {
            v4f x = s4[i];
            lmax = fmaxf(lmax, fmaxf(fmaxf(x[0], x[1]), fmaxf(x[2], x[3])));
        }
        red[row][ci] = lmax;
        __syncthreads();
        if (ci == 0) {
            float m = red[row][0];
#pragma unroll
            for (int i = 1; i < 16; ++i) m = fmaxf(m, red[row][i]);
            rmax_s[row] = m;
        }
        __syncthreads();
        float rm = rmax_s[row];
        float lsum = 0.f;
#pragma unroll 8
        for (int i = 0; i < 32; ++i) {
            v4f x = s4[i];
            lsum += __expf(x[0] - rm) + __expf(x[1] - rm)
                  + __expf(x[2] - rm) + __expf(x[3] - rm);
        }
        red[row][ci] = lsum;
        __syncthreads();
        if (ci == 0) {
            float s = 0.f;
#pragma unroll
            for (int i = 0; i < 16; ++i) s += red[row][i];
            rinv_s[row] = 1.0f / s;
        }
        __syncthreads();
    }

    // ---- phase B: P(16x32 f16) @ V(32x64), 8 chunks per wave -----------
    const int wave = tid >> 5, lane = tid & 31;
    const int lm = lane & 15, hl = lane >> 4;
    const float rm = rmax_s[lm];
    const float ri = rinv_s[lm];
    const float* prow = scores + ((size_t)bh * SEQ + q0 + lm) * SEQ + hl * 8;
    const _Float16* vb = vT + (size_t)bh * HDIM * SEQ + (size_t)lm * SEQ + hl * 8;

    v8f acc[4] = {};
    for (int c = wave; c < SEQ / 32; c += 8) {
        const int kk0 = c * 32;
        const v4f* p4 = (const v4f*)(prow + kk0);
        v4f x0 = p4[0], x1 = p4[1], x2 = p4[4], x3 = p4[5];
        const _Float16* vkk = vb + kk0;
        v16h bcur = gather16_f16(vkk);           // t=0
        v16h a;
#pragma unroll
        for (int i = 0; i < 4; ++i) {            // exp/convert fills MMA gaps
            a[i]      = (_Float16)(__expf(x0[i] - rm) * ri);
            a[4 + i]  = (_Float16)(__expf(x1[i] - rm) * ri);
            a[8 + i]  = (_Float16)(__expf(x2[i] - rm) * ri);
            a[12 + i] = (_Float16)(__expf(x3[i] - rm) * ri);
        }
#pragma unroll
        for (int t = 0; t < 4; ++t) {
            v16h bnxt = (t < 3) ? gather16_f16(vkk + (t + 1) * 16 * SEQ) : bcur;
            acc[t] = wmma_f16(a, bcur, acc[t]);
            bcur = bnxt;
        }
    }

#pragma unroll
    for (int t = 0; t < 4; ++t)
#pragma unroll
        for (int r = 0; r < 8; ++r)
            partial[wave][r + 8 * hl][t * 16 + lm] = acc[t][r];
    __syncthreads();

    // ---- cross-wave reduction + store f16 ------------------------------
    for (int i = tid; i < 16 * 64; i += 256) {
        int rr = i >> 6, dd = i & 63;
        float s = 0.f;
#pragma unroll
        for (int w = 0; w < 8; ++w) s += partial[w][rr][dd];
        size_t m = (size_t)b * SEQ + q0 + rr;
        oh[m * DMODEL + h * HDIM + dd] = (_Float16)s;
    }
}

// ======================================================================
// Kernel 4: out = oh16 @ Wo^T + residual   (f32 out), LDS-staged Wo slab
// ======================================================================
__global__ __launch_bounds__(256) void outproj_kernel(
    const _Float16* __restrict__ A, const float* __restrict__ W,
    const float* __restrict__ resid, float* __restrict__ out)
{
    __shared__ __align__(16) _Float16 sB[2][64 * BSTR];

    const int tid  = threadIdx.x;
    const int wave = tid >> 5, lane = tid & 31;
    const int lm = lane & 15, hl = lane >> 4;
    const int n0 = blockIdx.x * 64;
    const int m0 = blockIdx.y * 128 + wave * 16;

    const int wr = tid >> 2;
    const int wc = (tid & 3) * 8;
    const float* wsrc = W + (size_t)(n0 + wr) * DMODEL + wc;
    v4f f0, f1;

    const _Float16* arow = A + (size_t)(m0 + lm) * DMODEL + hl * 8;

    v8f acc[4] = {};

    f0 = *(const v4f*)(wsrc);  f1 = *(const v4f*)(wsrc + 4);
    {
        v8h h;
#pragma unroll
        for (int i = 0; i < 4; ++i) { h[i] = (_Float16)f0[i]; h[4 + i] = (_Float16)f1[i]; }
        *(v8h*)&sB[0][wr * BSTR + wc] = h;
    }
    __syncthreads();

    v16h a_cur = gather16_f16(arow);

#pragma unroll 2
    for (int kk = 0; kk < DMODEL; kk += 32) {
        const int cur = (kk >> 5) & 1;
        const bool last = (kk + 32 >= DMODEL);
        if (!last) {
            f0 = *(const v4f*)(wsrc + kk + 32);
            f1 = *(const v4f*)(wsrc + kk + 36);
            __builtin_prefetch(arow + kk + 64, 0, 1);
        }
        v16h a_nxt = last ? a_cur : gather16_f16(arow + kk + 32);

        v16h bf[4];
#pragma unroll
        for (int t = 0; t < 4; ++t)
            bf[t] = gather16_f16(&sB[cur][(t * 16 + lm) * BSTR + hl * 8]);
#pragma unroll
        for (int t = 0; t < 4; ++t)
            acc[t] = wmma_f16(a_cur, bf[t], acc[t]);

        if (!last) {
            __syncthreads();
            v8h h;
#pragma unroll
            for (int i = 0; i < 4; ++i) { h[i] = (_Float16)f0[i]; h[4 + i] = (_Float16)f1[i]; }
            *(v8h*)&sB[cur ^ 1][wr * BSTR + wc] = h;
            __syncthreads();
        }
        a_cur = a_nxt;
    }

#pragma unroll
    for (int t = 0; t < 4; ++t)
#pragma unroll
        for (int r = 0; r < 8; ++r) {
            size_t m = (size_t)m0 + r + 8 * hl;
            size_t n = (size_t)n0 + t * 16 + lm;
            out[m * DMODEL + n] = acc[t][r] + resid[m * DMODEL + n];
        }
}

// ======================================================================
extern "C" void kernel_launch(void* const* d_in, const int* in_sizes, int n_in,
                              void* d_out, int out_size, void* d_ws, size_t ws_size,
                              hipStream_t stream) {
    (void)in_sizes; (void)n_in; (void)out_size; (void)ws_size;
    const float* Q  = (const float*)d_in[0];
    const float* K  = (const float*)d_in[1];
    const float* V  = (const float*)d_in[2];
    const float* Wq = (const float*)d_in[3];
    const float* Wk = (const float*)d_in[4];
    const float* Wv = (const float*)d_in[5];
    const float* Wo = (const float*)d_in[6];
    const int*  mask = (const int*)d_in[7];

    float* out0 = (float*)d_out;                 // [B,S,D]
    float* scores = out0 + OUT0_ELEMS;           // [B,H,S,S] masked pre-softmax

    _Float16* q16 = (_Float16*)d_ws;             // [B,H,S,hd], pre-scaled
    _Float16* k16 = q16 + QKV_ELEMS;             // [B,H,S,hd]
    _Float16* vT  = k16 + QKV_ELEMS;             // [B,H,hd,S]
    _Float16* oh  = vT  + QKV_ELEMS;             // [B,S,D]

    const dim3 blk(256);
    const dim3 gProj(DMODEL / 64, MROWS / 128);          // (16, 32)

    const float qscale = 0.125f;                         // 1/sqrt(64)
    proj_kernel<0><<<gProj, blk, 0, stream>>>(Q, Wq, q16, qscale);
    proj_kernel<0><<<gProj, blk, 0, stream>>>(K, Wk, k16, 1.0f);
    proj_kernel<1><<<gProj, blk, 0, stream>>>(V, Wv, vT, 1.0f);

    score_kernel<<<dim3(2, SEQ / 16, BATCH * NHEAD), blk, 0, stream>>>(
        q16, k16, mask, scores);

    softmax_pv_kernel<<<dim3(SEQ / 16, BATCH * NHEAD), blk, 0, stream>>>(
        scores, vT, oh);

    outproj_kernel<<<gProj, blk, 0, stream>>>(oh, Wo, Q, out0);
}
